// GeneratorHead_87256555586282
// MI455X (gfx1250) — matrix operand order repaired and checked by reading, not compiled, not run on hardware
//
#include <hip/hip_runtime.h>

// ---------------------------------------------------------------------------
// KAN generator head for MI455X (gfx1250, wave32, WMMA).
//
// Fused formulation:  kan_linear(x) = [silu(x) | bspline8(x)] @ [Wb | Ws*scaler]^T
//   GEMM1: (2048 x 6912)  @ (256   x 6912)^T -> h1, + exact GELU epilogue
//   LN + rebuild activations -> A2 (2048 x 2304)
//   GEMM2: (2048 x 2304)  @ (30528 x 2304)^T -> logits (guard N<30522)
//
// Precision: bf16 operands + f32 WMMA accumulate (v_wmma_f32_16x16x32_bf16).
// bf16 W2 (141 MB) fits in the 192 MB L2 -> the big GEMM is L2-resident and
// compute-bound on the bf16 WMMA pipe.
//
// GEMM staging: CDNA5 async global->LDS copies (GLOBAL_LOAD_ASYNC_TO_LDS_B128,
// ASYNCcnt) into double-buffered LDS -- zero staging VGPRs, so no scratch
// spills; next-stage copies overlap the WMMA chain of the current stage.
// ---------------------------------------------------------------------------

typedef __attribute__((ext_vector_type(16))) __bf16 v16bf;
typedef __attribute__((ext_vector_type(8)))  __bf16 v8bf;
typedef __attribute__((ext_vector_type(8)))  float  v8f;
typedef __attribute__((ext_vector_type(4)))  int    v4i;

#define NROWS   2048
#define HDIM    768
#define EDIM    256
#define VDIM    30522
#define VPAD    30528          // padded to multiple of 64
#define K1      (HDIM * 9)     // 6912 = 768 silu + 768*8 spline bases
#define K2      (EDIM * 9)     // 2304 = 256 silu + 256*8 spline bases

#if __has_builtin(__builtin_amdgcn_global_load_async_to_lds_b128)
#define HAVE_ASYNC_LDS 1
#endif

// 16-byte global -> LDS copy. Async path uses no VGPR data registers.
// Builtin signature (from hipcc diagnostic): (v4i AS1*, v4i AS3*, imm, imm)
__device__ __forceinline__ void cp16_g2l(const unsigned short* g, unsigned short* l) {
#ifdef HAVE_ASYNC_LDS
    __builtin_amdgcn_global_load_async_to_lds_b128(
        (__attribute__((address_space(1))) v4i*)g,
        (__attribute__((address_space(3))) v4i*)l, 0, 0);
#else
    *(uint4*)l = *(const uint4*)g;
#endif
}

__device__ __forceinline__ void wait_async0() {
#ifdef HAVE_ASYNC_LDS
#if __has_builtin(__builtin_amdgcn_s_wait_asynccnt)
    __builtin_amdgcn_s_wait_asynccnt(0);
#else
    asm volatile("s_wait_asynccnt 0x0" ::: "memory");
#endif
#endif
}

// round-to-nearest-even f32 -> bf16 (bit pattern in ushort)
__device__ __forceinline__ unsigned short f2bf(float f) {
    unsigned int u = __float_as_uint(f);
    unsigned int r = u + 0x7FFFu + ((u >> 16) & 1u);
    return (unsigned short)(r >> 16);
}

__device__ __forceinline__ float silu_f(float x) {
    return x / (1.0f + expf(-x));
}

// cubic B-spline bases on uniform grid: knots t_i = -1 + (i-3)*0.4, i=0..11
// (GRID_SIZE=5, SPLINE_ORDER=3 -> 8 basis values), Cox-de-Boor recursion.
__device__ __forceinline__ void bspline8(float x, float out[8]) {
    const float h = 0.4f;
    float t[12];
#pragma unroll
    for (int i = 0; i < 12; ++i) t[i] = -1.0f + (float)(i - 3) * h;
    float b[11];
#pragma unroll
    for (int i = 0; i < 11; ++i) b[i] = (x >= t[i] && x < t[i + 1]) ? 1.0f : 0.0f;
#pragma unroll
    for (int p = 1; p <= 3; ++p) {
        float ph = (float)p * h;
#pragma unroll
        for (int i = 0; i < 10; ++i) {
            if (i < 11 - p) {
                float left  = (x - t[i]) / ph;
                float right = (t[i + p + 1] - x) / ph;
                b[i] = left * b[i] + right * b[i + 1];
            }
        }
    }
#pragma unroll
    for (int i = 0; i < 8; ++i) out[i] = b[i];
}

// ---------------------------------------------------------------------------
// Pack kernels: bf16 augmented weights / activations
// ---------------------------------------------------------------------------
__global__ void build_w1(const float* __restrict__ bw, const float* __restrict__ sw,
                         const float* __restrict__ sc, unsigned short* __restrict__ W) {
    int idx = blockIdx.x * 256 + threadIdx.x;
    const int total = EDIM * K1;
    if (idx >= total) return;
    int o = idx / K1, c = idx - o * K1;
    float v;
    if (c < HDIM) {
        v = bw[o * HDIM + c];
    } else {
        int cc = c - HDIM, i = cc >> 3, j = cc & 7;
        v = sw[(o * HDIM + i) * 8 + j] * sc[o * HDIM + i];
    }
    W[idx] = f2bf(v);
}

__global__ void build_w2(const float* __restrict__ bw, const float* __restrict__ sw,
                         const float* __restrict__ sc, unsigned short* __restrict__ W) {
    int idx = blockIdx.x * 256 + threadIdx.x;
    const int total = VPAD * K2;
    if (idx >= total) return;
    int o = idx / K2, c = idx - o * K2;
    float v = 0.0f;
    if (o < VDIM) {
        if (c < EDIM) {
            v = bw[o * EDIM + c];
        } else {
            int cc = c - EDIM, i = cc >> 3, j = cc & 7;
            v = sw[(o * EDIM + i) * 8 + j] * sc[o * EDIM + i];
        }
    }
    W[idx] = f2bf(v);
}

__global__ void build_a1(const float* __restrict__ hidden, unsigned short* __restrict__ A1) {
    int idx = blockIdx.x * 256 + threadIdx.x;
    const int total = NROWS * HDIM;
    if (idx >= total) return;
    int n = idx / HDIM, i = idx - n * HDIM;
    float x = hidden[idx];
    unsigned short* row = A1 + (size_t)n * K1;
    row[i] = f2bf(silu_f(x));
    float b[8];
    bspline8(x, b);
#pragma unroll
    for (int j = 0; j < 8; ++j) row[HDIM + i * 8 + j] = f2bf(b[j]);
}

// ---------------------------------------------------------------------------
// LayerNorm (eps=1e-12) over E=256 + build augmented A2 row (bf16)
// one 256-thread block per row
// ---------------------------------------------------------------------------
__global__ void ln_build_a2(const float* __restrict__ h1, const float* __restrict__ g,
                            const float* __restrict__ be, unsigned short* __restrict__ A2) {
    int n = blockIdx.x, e = threadIdx.x;
    float y = h1[(size_t)n * EDIM + e];
    __shared__ float red[EDIM];
    red[e] = y;
    __syncthreads();
    for (int s = 128; s > 0; s >>= 1) {
        if (e < s) red[e] += red[e + s];
        __syncthreads();
    }
    float mu = red[0] * (1.0f / EDIM);
    __syncthreads();
    float d = y - mu;
    red[e] = d * d;
    __syncthreads();
    for (int s = 128; s > 0; s >>= 1) {
        if (e < s) red[e] += red[e + s];
        __syncthreads();
    }
    float var = red[0] * (1.0f / EDIM);
    float yn = d * rsqrtf(var + 1e-12f) * g[e] + be[e];
    unsigned short* row = A2 + (size_t)n * K2;
    row[e] = f2bf(silu_f(yn));
    float b[8];
    bspline8(yn, b);
#pragma unroll
    for (int j = 0; j < 8; ++j) row[EDIM + e * 8 + j] = f2bf(b[j]);
}

// ---------------------------------------------------------------------------
// Tiled bf16 WMMA GEMM:  C[M x n_real] = A[M x K] * W[Npad x K]^T  (f32 acc)
//
// Block: 256 threads = 8 wave32s. Block tile (WM*128)M x 64N, K-step 64.
// Each wave owns WM*16 M-rows x 64 N-cols: WM x 4 accumulators,
// WM*4 WMMAs per 16x16x32 K-step.
//
// Staging: per thread WM*4+2 async b128 copies per stage into the alternate
// LDS buffer, issued right after the barrier so they overlap the WMMA chain.
//   s_wait_asynccnt 0 ; barrier ; issue(next) ; compute(cur)
//
// Fragment layouts used here (16-bit, wave32; A/B use contiguous k, same order in both):
//   A 16x32: lane(m=l&15, half=l>>4): elems 0..15 = K[half*16..+16)
//   B 32x16: lane(n=l&15, half=l>>4): elems 0..15 = K[half*16..+16)
//   C 16x16: VGPR r -> M = r + 8*half, N = lane&15
// ---------------------------------------------------------------------------
template <bool GELU, int WM>
__global__ __launch_bounds__(256, 1) void kan_gemm(const unsigned short* __restrict__ A,
                                                   const unsigned short* __restrict__ W,
                                                   float* __restrict__ C,
                                                   int K, int ldc, int n_real) {
    __shared__ __align__(16) unsigned short As[2][WM * 128 * 64];
    __shared__ __align__(16) unsigned short Bs[2][64 * 64];

    const int tid  = threadIdx.x;
    const int lane = tid & 31;
    const int wave = tid >> 5;
    const int m    = lane & 15;   // also serves as n for B fragments
    const int half = lane >> 4;
    const int blockM = blockIdx.y * (WM * 128);
    const int blockN = blockIdx.x * 64;

    v8f acc[WM][4] = {};

    // issue one stage of global->LDS copies (no data registers held)
    auto stage_issue = [&](int k0, int buf) {
#pragma unroll
        for (int i = 0; i < WM * 4; ++i) {
            int chunk = tid + i * 256;            // (WM*128 rows) * 8 chunks
            int row = chunk >> 3, cc = chunk & 7;
            cp16_g2l(&A[(size_t)(blockM + row) * K + k0 + cc * 8],
                     &As[buf][row * 64 + cc * 8]);
        }
#pragma unroll
        for (int i = 0; i < 2; ++i) {
            int chunk = tid + i * 256;            // 64 rows * 8 chunks
            int row = chunk >> 3, cc = chunk & 7;
            cp16_g2l(&W[(size_t)(blockN + row) * K + k0 + cc * 8],
                     &Bs[buf][row * 64 + cc * 8]);
        }
    };

    // prologue: stage 0
    stage_issue(0, 0);

    int cur = 0;
    for (int k0 = 0; k0 < K; k0 += 64) {
        wait_async0();                         // this wave's copies into buf[cur] done
        __syncthreads();                       // all waves done; buf[cur^1] free
        const bool has_next = (k0 + 64) < K;
        if (has_next) {
            stage_issue(k0 + 64, cur ^ 1);     // overlaps the WMMA chain below
            __builtin_prefetch((const void*)&W[(size_t)(blockN + (tid >> 2)) * K +
                                               k0 + 128 + (tid & 3) * 16], 0, 3);
        }

        // ---- two 16x16x32 WMMA K-steps on buf[cur] ----
#pragma unroll
        for (int kk = 0; kk < 64; kk += 32) {
            union { v16bf v; v8bf h[2]; } af[WM];
#pragma unroll
            for (int mm = 0; mm < WM; ++mm) {
                const unsigned short* arow = &As[cur][((wave * WM + mm) * 16 + m) * 64 + kk];
                af[mm].v = *(const v16bf*)(arow + half * 16);
            }
            v16bf bv[4];
#pragma unroll
            for (int t = 0; t < 4; ++t) {
                bv[t] = *(const v16bf*)(&Bs[cur][(t * 16 + m) * 64 + kk + half * 16]);
            }
#pragma unroll
            for (int t = 0; t < 4; ++t) {
#pragma unroll
                for (int mm = 0; mm < WM; ++mm) {
                    acc[mm][t] = __builtin_amdgcn_wmma_f32_16x16x32_bf16(
                        false, af[mm].v, false, bv[t], (short)0, acc[mm][t], false, false);
                }
            }
        }
        cur ^= 1;
    }

    // ---- epilogue ----
#pragma unroll
    for (int mm = 0; mm < WM; ++mm) {
        const int rowBase = blockM + (wave * WM + mm) * 16 + half * 8;
#pragma unroll
        for (int t = 0; t < 4; ++t) {
            int col = blockN + t * 16 + m;
            if (col < n_real) {
#pragma unroll
                for (int r = 0; r < 8; ++r) {
                    float v = acc[mm][t][r];
                    if (GELU) v = 0.5f * v * (1.0f + erff(v * 0.70710678118654752f));
                    C[(size_t)(rowBase + r) * ldc + col] = v;
                }
            }
        }
    }
}

// ---------------------------------------------------------------------------
// Launch: workspace layout:
//   W2 bf16 : VPAD*K2  = 70,336,512 elems
//   A1 bf16 : NROWS*K1 = 14,155,776 elems
//   W1 bf16 : EDIM*K1  =  1,769,472 elems
//   A2 bf16 : NROWS*K2 =  4,718,592 elems
//   H1 f32  : NROWS*EDIM =  524,288 elems     (total ~184 MB)
// ---------------------------------------------------------------------------
extern "C" void kernel_launch(void* const* d_in, const int* in_sizes, int n_in,
                              void* d_out, int out_size, void* d_ws, size_t ws_size,
                              hipStream_t stream) {
    const float* hidden    = (const float*)d_in[0];
    const float* base_w1   = (const float*)d_in[1];
    const float* spline_w1 = (const float*)d_in[2];
    const float* scaler1   = (const float*)d_in[3];
    /* d_in[4] = grid1 (uniform, hardcoded) */
    const float* ln_g      = (const float*)d_in[5];
    const float* ln_b      = (const float*)d_in[6];
    const float* base_w2   = (const float*)d_in[7];
    const float* spline_w2 = (const float*)d_in[8];
    const float* scaler2   = (const float*)d_in[9];
    /* d_in[10] = grid2 (uniform, hardcoded) */
    (void)in_sizes; (void)n_in; (void)out_size; (void)ws_size;

    unsigned short* W2 = (unsigned short*)d_ws;
    unsigned short* A1 = W2 + (size_t)VPAD * K2;
    unsigned short* W1 = A1 + (size_t)NROWS * K1;
    unsigned short* A2 = W1 + (size_t)EDIM * K1;
    float*          H1 = (float*)(A2 + (size_t)NROWS * K2);

    // Pack bf16 weights & layer-1 activations
    {
        int total = VPAD * K2;
        build_w2<<<(total + 255) / 256, 256, 0, stream>>>(base_w2, spline_w2, scaler2, W2);
    }
    {
        int total = EDIM * K1;
        build_w1<<<(total + 255) / 256, 256, 0, stream>>>(base_w1, spline_w1, scaler1, W1);
    }
    {
        int total = NROWS * HDIM;
        build_a1<<<(total + 255) / 256, 256, 0, stream>>>(hidden, A1);
    }

    // Layer 1 GEMM (+GELU epilogue): (2048 x 6912) x (256 x 6912)^T, BM=128
    kan_gemm<true, 1><<<dim3(EDIM / 64, NROWS / 128), 256, 0, stream>>>(
        A1, W1, H1, K1, EDIM, EDIM);

    // LayerNorm + layer-2 activation build
    ln_build_a2<<<NROWS, 256, 0, stream>>>(H1, ln_g, ln_b, A2);

    // Layer 2 GEMM: (2048 x 2304) x (30528 x 2304)^T -> logits, BM=256
    kan_gemm<false, 2><<<dim3(VPAD / 64, NROWS / 256), 256, 0, stream>>>(
        A2, W2, (float*)d_out, K2, VDIM, VDIM);
}
